// TemporalGAT_90288802496597
// MI455X (gfx1250) — matrix operand order repaired and checked by reading, not compiled, not run on hardware
//
#include <hip/hip_runtime.h>
#include <hip/hip_bf16.h>

#define GAT_ALPHA 0.2f

constexpr int Bm  = 16;
constexpr int Lm  = 1024;
constexpr int Fm  = 128;
constexpr int Hm  = 4;
constexpr int Dm  = 64;
constexpr int HDm = 256;

typedef __attribute__((ext_vector_type(16))) __bf16 v16bf;
typedef __attribute__((ext_vector_type(8)))  __bf16 v8bf;
typedef __attribute__((ext_vector_type(8)))  float  v8f;
typedef __attribute__((ext_vector_type(4)))  unsigned int v4u;
typedef __attribute__((ext_vector_type(8)))  int v8i;
typedef __attribute__((ext_vector_type(4)))  int v4i;

#if __has_builtin(__builtin_amdgcn_tensor_load_to_lds)
#define HAVE_TDM 1
#else
#define HAVE_TDM 0
#endif

__device__ __forceinline__ v8f zero8() {
    v8f z;
#pragma unroll
    for (int i = 0; i < 8; ++i) z[i] = 0.0f;
    return z;
}

__device__ __forceinline__ v8f wmma_bf16(v16bf a, v16bf b, v8f c) {
    // D = A(16x32 bf16) x B(32x16 bf16) + C(16x16 f32)
    return __builtin_amdgcn_wmma_f32_16x16x32_bf16(
        false, a, false, b, (short)0, c, false, false);
}

// A-matrix (16x32, 16-bit): lane holds row M = lane&15; halves 0..7 = K kb..kb+7,
// halves 8..15 = K kb+16..kb+23, kb = (lane>>4)*8.
__device__ __forceinline__ v16bf load_a(const __bf16* p, int lda, int lane) {
    const int r  = lane & 15;
    const int kb = (lane >> 4) << 3;
    const __bf16* q = p + r * lda + kb;
    const v8bf lo = *(const v8bf*)(q);
    const v8bf hi = *(const v8bf*)(q + 16);
    v16bf a;
#pragma unroll
    for (int t = 0; t < 8; ++t) { a[t] = lo[t]; a[8 + t] = hi[t]; }
    return a;
}

// B-matrix (32x16) staged TRANSPOSED in LDS: bt[n][k], row stride ldb halves
// (ldb*2 must be multiple of 16B). Lane: column N = lane&15, K = kb..kb+7, kb+16..kb+23, kb = (lane>>4)*8.
__device__ __forceinline__ v16bf load_bt(const __bf16* p, int ldb, int lane) {
    const int n  = lane & 15;
    const int kb = (lane >> 4) << 3;
    const __bf16* q = p + n * ldb + kb;
    const v8bf x = *(const v8bf*)(q);
    const v8bf y = *(const v8bf*)(q + 16);
    v16bf b;
#pragma unroll
    for (int t = 0; t < 8; ++t) { b[t] = x[t]; b[8 + t] = y[t]; }
    return b;
}

#if HAVE_TDM
// ---- Tensor Data Mover: 2D bf16 tile (global -> LDS), optional row padding.
// D# group0/group1 packing per CDNA5 ISA 8.3/8.4.
struct TdmDesc { v4u g0; v8i g1; };

__device__ __forceinline__ TdmDesc make_tdm_2d_bf16(
    const void* gptr, unsigned lds_off, unsigned tile_w, unsigned tile_h,
    unsigned row_stride_elems, bool pad_en, unsigned pad_interval_code,
    unsigned pad_amount_code) {
    const unsigned long long ga = (unsigned long long)(size_t)gptr;
    TdmDesc d;
    d.g0[0] = 1u;                                   // count=1, user-mode
    d.g0[1] = lds_off;                              // LDS byte address
    d.g0[2] = (unsigned)(ga & 0xffffffffu);         // global_addr[31:0]
    d.g0[3] = (unsigned)((ga >> 32) & 0x01ffffffu)  // global_addr[56:32]
              | 0x80000000u;                        // type=2 ("image")
    const unsigned tw = 0x7fffffffu;                // huge tensor dims: no OOB clip
    d.g1[0] = (int)((1u << 16)                      // data_size=1 (2 bytes)
              | (pad_en ? (1u << 20) : 0u)
              | (pad_interval_code << 22)
              | (pad_amount_code << 25));
    d.g1[1] = (int)((tw & 0xffffu) << 16);                  // tensor_dim0[15:0]
    d.g1[2] = (int)((tw >> 16) | ((tw & 0xffffu) << 16));   // dim0 hi | dim1 lo
    d.g1[3] = (int)((tw >> 16) | (tile_w << 16));           // dim1 hi | tile_dim0
    d.g1[4] = (int)(tile_h & 0xffffu);                      // tile_dim1 (tile_dim2=0)
    d.g1[5] = (int)row_stride_elems;                        // dim0_stride[31:0]
    d.g1[6] = 0;
    d.g1[7] = 0;
    return d;
}

__device__ __forceinline__ void tdm_load(const TdmDesc& d) {
    v4i z4; v8i z8;
#pragma unroll
    for (int i = 0; i < 4; ++i) z4[i] = 0;
#pragma unroll
    for (int i = 0; i < 8; ++i) z8[i] = 0;
#if defined(__clang_major__) && (__clang_major__ >= 23)
    __builtin_amdgcn_tensor_load_to_lds(d.g0, d.g1, z4, z4, z8, 0);
#else
    __builtin_amdgcn_tensor_load_to_lds(d.g0, d.g1, z4, z4, 0);
#endif
}

__device__ __forceinline__ unsigned lds_off_of(const void* p) {
    return (unsigned)(size_t)p;  // low 32 bits of flat shared addr = LDS offset
}
#endif  // HAVE_TDM

// ---------------------------------------------------------------------------
// Kernel 1: Wh = x @ W[h]; writes WhbT (bf16, [H,B,D,L] transposed) + si/sj.
// ---------------------------------------------------------------------------
__global__ __launch_bounds__(256) void k_wh(
    const float* __restrict__ x, const float* __restrict__ W,
    const float* __restrict__ a1, const float* __restrict__ a2,
    __bf16* __restrict__ WhbT, float* __restrict__ si, float* __restrict__ sj) {
    __shared__ __attribute__((aligned(16))) char smem[32768 + 64 * 136 * 2];
    __bf16* sA  = (__bf16*)smem;            // [128][128] bf16 x-tile
    __bf16* sBT = (__bf16*)(smem + 32768);  // [64][136]  W[h]^T (padded)
    float*  sWh = (float*)smem;             // [128][64]  (reuses sA region)

    const int tid = threadIdx.x, lane = tid & 31, wv = tid >> 5;
    const int h   = blockIdx.x / (Bm * 8);
    const int rem = blockIdx.x % (Bm * 8);
    const int b   = rem >> 3;
    const int l0  = (rem & 7) * 128;

    const float* xp = x + ((size_t)b * Lm + l0) * Fm;
    for (int idx = tid; idx < 128 * 128; idx += 256) sA[idx] = (__bf16)xp[idx];

    const float* wp = W + (size_t)h * Fm * Dm;
    for (int idx = tid; idx < 128 * 64; idx += 256) {
        const int f = idx >> 6, d = idx & 63;
        sBT[d * 136 + f] = (__bf16)wp[f * Dm + d];
    }
    __syncthreads();

    v8f acc[4];
#pragma unroll
    for (int i = 0; i < 4; ++i) acc[i] = zero8();
    const __bf16* Arow = sA + wv * 16 * 128;
#pragma unroll
    for (int kk = 0; kk < 128; kk += 32) {
        const v16bf a = load_a(Arow + kk, 128, lane);
#pragma unroll
        for (int nt = 0; nt < 4; ++nt) {
            const v16bf bm = load_bt(sBT + nt * 16 * 136 + kk, 136, lane);
            acc[nt] = wmma_bf16(a, bm, acc[nt]);
        }
    }
    __syncthreads();  // done reading sA; reuse region as sWh

    const int n = lane & 15, mb = (lane >> 4) << 3;
#pragma unroll
    for (int nt = 0; nt < 4; ++nt)
#pragma unroll
        for (int r = 0; r < 8; ++r)
            sWh[(wv * 16 + mb + r) * 64 + nt * 16 + n] = acc[nt][r];
    __syncthreads();

    // Wh transposed: WhbT[hb, d, l]  (rows contiguous in l for TDM tiles).
    __bf16* whtp = WhbT + ((size_t)h * Bm + b) * Dm * Lm;
    for (int idx = tid; idx < 128 * 64; idx += 256) {
        const int d = idx >> 7, l = idx & 127;
        whtp[(size_t)d * Lm + l0 + l] = (__bf16)sWh[l * 64 + d];
    }

    if (tid < 128) {
        const float* A1 = a1 + h * Dm;
        const float* A2 = a2 + h * Dm;
        float s1 = 0.0f, s2 = 0.0f;
#pragma unroll 8
        for (int d = 0; d < Dm; ++d) {
            const float v = sWh[tid * 64 + d];
            s1 += v * A1[d];
            s2 += v * A2[d];
        }
        const size_t o = ((size_t)h * Bm + b) * Lm + l0 + tid;
        si[o] = s1;
        sj[o] = s2;
    }
}

// ---------------------------------------------------------------------------
// Kernel 2: per-row softmax stats for e_ij = lrelu(si_i + sj_j).
// ---------------------------------------------------------------------------
__global__ __launch_bounds__(256) void k_gat_stats(
    const float* __restrict__ si, const float* __restrict__ sj,
    float* __restrict__ gm, float* __restrict__ ginv) {
    __shared__ float ssj[Lm];
    const size_t hb = blockIdx.x;
    for (int j = threadIdx.x; j < Lm; j += 256) ssj[j] = sj[hb * Lm + j];
    __syncthreads();
    for (int i = threadIdx.x; i < Lm; i += 256) {
        const float v = si[hb * Lm + i];
        float m = -3.4e38f;
        for (int j = 0; j < Lm; ++j) {
            const float t = v + ssj[j];
            m = fmaxf(m, t > 0.0f ? t : GAT_ALPHA * t);
        }
        float s = 0.0f;
        for (int j = 0; j < Lm; ++j) {
            const float t = v + ssj[j];
            const float e = t > 0.0f ? t : GAT_ALPHA * t;
            s += __expf(e - m);
        }
        gm[hb * Lm + i]   = m;
        ginv[hb * Lm + i] = 1.0f / s;
    }
}

// ---------------------------------------------------------------------------
// Kernel 3: stream gat_att out (NT stores) and h = att @ Wh via WMMA.
// Wh B-tiles: double-buffered TDM DMAs, in flight across softmax + WMMA.
// ---------------------------------------------------------------------------
__global__ __launch_bounds__(256) void k_gat_att(
    const float* __restrict__ si, const float* __restrict__ sj,
    const float* __restrict__ gm, const float* __restrict__ ginv,
    const __bf16* __restrict__ WhbT,
    float* __restrict__ gat_att, float* __restrict__ gat_out) {
    __shared__ __attribute__((aligned(16))) __bf16 sAtt[128 * 32];
    __shared__ __attribute__((aligned(16))) __bf16 sWT[2][64 * 40];
    __shared__ float s_si[128], s_m[128], s_inv[128], s_sj[Lm];

    const int tid = threadIdx.x, lane = tid & 31, wv = tid >> 5;
    const int h   = blockIdx.x / (Bm * 8);
    const int rem = blockIdx.x % (Bm * 8);
    const int b   = rem >> 3;
    const int l0  = (rem & 7) * 128;
    const size_t hb = (size_t)h * Bm + b;

    for (int j = tid; j < Lm; j += 256) s_sj[j] = sj[hb * Lm + j];
    if (tid < 128) {
        s_si[tid]  = si[hb * Lm + l0 + tid];
        s_m[tid]   = gm[hb * Lm + l0 + tid];
        s_inv[tid] = ginv[hb * Lm + l0 + tid];
    }
    v8f acc[4];
#pragma unroll
    for (int i = 0; i < 4; ++i) acc[i] = zero8();

    float* attp       = gat_att + (hb * Lm + l0) * Lm;
    const __bf16* whT = WhbT + hb * Dm * Lm;  // [64][1024]
    __syncthreads();

#if HAVE_TDM
    if (wv == 0)  // prologue: tile j0=0 into buffer 0
        tdm_load(make_tdm_2d_bf16(whT, lds_off_of(sWT[0]), 32, 64, Lm,
                                  true, 3, 3));  // pad 64B rows -> 80B
#endif
    for (int j0 = 0; j0 < Lm; j0 += 32) {
        const int cur = (j0 >> 5) & 1, nxt = cur ^ 1;
#if HAVE_TDM
        if (wv == 0 && j0 + 32 < Lm)  // next tile; prev reads done (barrier)
            tdm_load(make_tdm_2d_bf16(whT + j0 + 32, lds_off_of(sWT[nxt]),
                                      32, 64, Lm, true, 3, 3));
#endif
        for (int idx = tid; idx < 128 * 32; idx += 256) {
            const int i = idx >> 5, j = idx & 31;
            const float t = s_si[i] + s_sj[j0 + j];
            const float e = t > 0.0f ? t : GAT_ALPHA * t;
            const float p = __expf(e - s_m[i]) * s_inv[i];
            __builtin_nontemporal_store(p, &attp[(size_t)i * Lm + j0 + j]);
            sAtt[idx] = (__bf16)p;
        }
#if HAVE_TDM
        if (wv == 0) {
            if (j0 + 32 < Lm)
                __builtin_amdgcn_s_wait_tensorcnt(1);  // cur done, nxt in flight
            else
                __builtin_amdgcn_s_wait_tensorcnt(0);
        }
#else
        for (int idx = tid; idx < 64 * 32; idx += 256) {
            const int d = idx >> 5, jj = idx & 31;
            sWT[cur][d * 40 + jj] = whT[(size_t)d * Lm + j0 + jj];
        }
#endif
        __syncthreads();
        const v16bf a = load_a(sAtt + wv * 16 * 32, 32, lane);
#pragma unroll
        for (int nt = 0; nt < 4; ++nt) {
            const v16bf bm = load_bt(sWT[cur] + nt * 16 * 40, 40, lane);
            acc[nt] = wmma_bf16(a, bm, acc[nt]);
        }
        __syncthreads();
    }

    const int n = lane & 15, mb = (lane >> 4) << 3;
    float* gop = gat_out + ((size_t)b * Lm + l0 + wv * 16) * HDm + h * Dm;
#pragma unroll
    for (int nt = 0; nt < 4; ++nt)
#pragma unroll
        for (int r = 0; r < 8; ++r)
            gop[(size_t)(mb + r) * HDm + nt * 16 + n] = acc[nt][r];
}

// ---------------------------------------------------------------------------
// Kernel 4: Q/K/V projections.  z==0/1 store row-major bf16 (Qb/Kb);
// z==2 stores V transposed (VbT [B,256,L]) for TDM B-tiles in att@V.
// ---------------------------------------------------------------------------
__global__ __launch_bounds__(256) void k_qkv(
    const float* __restrict__ go,
    const float* __restrict__ Wq, const float* __restrict__ bq,
    const float* __restrict__ Wk, const float* __restrict__ bk,
    const float* __restrict__ Wv, const float* __restrict__ bv,
    __bf16* __restrict__ Qb, __bf16* __restrict__ Kb,
    __bf16* __restrict__ VbT) {
    const float* Wm = Wq; const float* bias = bq;
    if (blockIdx.z == 1) { Wm = Wk; bias = bk; }
    else if (blockIdx.z == 2) { Wm = Wv; bias = bv; }

    __shared__ __attribute__((aligned(16))) __bf16 sA[128 * 32];
    __shared__ __attribute__((aligned(16))) __bf16 sBT[64 * 40];
    const int tid = threadIdx.x, lane = tid & 31, wv = tid >> 5;
    const int r0 = (blockIdx.x >> 2) * 128;
    const int n0 = (blockIdx.x & 3) * 64;

    v8f acc[4];
#pragma unroll
    for (int i = 0; i < 4; ++i) acc[i] = zero8();

    for (int k0 = 0; k0 < HDm; k0 += 32) {
        for (int idx = tid; idx < 128 * 32; idx += 256) {
            const int i = idx >> 5, kk = idx & 31;
            sA[idx] = (__bf16)go[(size_t)(r0 + i) * HDm + k0 + kk];
        }
        for (int idx = tid; idx < 64 * 32; idx += 256) {
            const int nn = idx & 63, kk = idx >> 6;
            sBT[nn * 40 + kk] = (__bf16)Wm[(size_t)(k0 + kk) * HDm + n0 + nn];
        }
        __syncthreads();
        const v16bf a = load_a(sA + wv * 16 * 32, 32, lane);
#pragma unroll
        for (int nt = 0; nt < 4; ++nt) {
            const v16bf bm = load_bt(sBT + nt * 16 * 40, 40, lane);
            acc[nt] = wmma_bf16(a, bm, acc[nt]);
        }
        __syncthreads();
    }
    const int n = lane & 15, mb = (lane >> 4) << 3;
    if (blockIdx.z == 2) {
        const int bb = r0 / Lm, il = r0 % Lm;  // 128-row tiles never span b
#pragma unroll
        for (int nt = 0; nt < 4; ++nt) {
            const float bn = bias[n0 + nt * 16 + n];
#pragma unroll
            for (int r = 0; r < 8; ++r)
                VbT[((size_t)bb * HDm + n0 + nt * 16 + n) * Lm + il + wv * 16 +
                    mb + r] = (__bf16)(acc[nt][r] + bn);
        }
    } else {
        __bf16* dst = (blockIdx.z == 0) ? Qb : Kb;
#pragma unroll
        for (int nt = 0; nt < 4; ++nt) {
            const float bn = bias[n0 + nt * 16 + n];
#pragma unroll
            for (int r = 0; r < 8; ++r)
                dst[(size_t)(r0 + wv * 16 + mb + r) * HDm + n0 + nt * 16 + n] =
                    (__bf16)(acc[nt][r] + bn);
        }
    }
}

// ---------------------------------------------------------------------------
// Kernel 5: raw scores = (Q @ K^T) * scale into the t_att slot.
// Q/K tiles double-buffered TDM: next k-step's DMAs overlap this step's WMMA.
// ---------------------------------------------------------------------------
__global__ __launch_bounds__(256) void k_scores(
    const __bf16* __restrict__ Qb, const __bf16* __restrict__ Kb,
    float* __restrict__ tatt) {
    __shared__ __attribute__((aligned(16))) __bf16 sA[2][128 * 32];
    __shared__ __attribute__((aligned(16))) __bf16 sBT[2][64 * 40];
    const int tid = threadIdx.x, lane = tid & 31, wv = tid >> 5;
    const int b   = blockIdx.x / 128;
    const int rem = blockIdx.x % 128;
    const int i0  = (rem >> 4) * 128;
    const int j0  = (rem & 15) * 64;

    const __bf16* qp = Qb + ((size_t)b * Lm + i0) * HDm;
    const __bf16* kp = Kb + ((size_t)b * Lm + j0) * HDm;

    v8f acc[4];
#pragma unroll
    for (int i = 0; i < 4; ++i) acc[i] = zero8();

#if HAVE_TDM
    if (wv == 0) {  // prologue: k0=0 tiles into buffer 0
        tdm_load(make_tdm_2d_bf16(qp, lds_off_of(sA[0]), 32, 128, HDm,
                                  false, 0, 0));
        tdm_load(make_tdm_2d_bf16(kp, lds_off_of(sBT[0]), 32, 64, HDm,
                                  true, 3, 3));
    }
#endif
    for (int step = 0; step < HDm / 32; ++step) {
        const int cur = step & 1, nxt = cur ^ 1;
        const int k0 = step * 32;
#if HAVE_TDM
        if (wv == 0) {
            if (step + 1 < HDm / 32) {
                tdm_load(make_tdm_2d_bf16(qp + k0 + 32, lds_off_of(sA[nxt]),
                                          32, 128, HDm, false, 0, 0));
                tdm_load(make_tdm_2d_bf16(kp + k0 + 32, lds_off_of(sBT[nxt]),
                                          32, 64, HDm, true, 3, 3));
                __builtin_amdgcn_s_wait_tensorcnt(2);  // cur pair retired
            } else {
                __builtin_amdgcn_s_wait_tensorcnt(0);
            }
        }
#else
        for (int idx = tid; idx < 128 * 32; idx += 256) {
            const int i = idx >> 5, kk = idx & 31;
            sA[cur][idx] = qp[(size_t)i * HDm + k0 + kk];
        }
        for (int idx = tid; idx < 64 * 32; idx += 256) {
            const int nn = idx >> 5, kk = idx & 31;
            sBT[cur][nn * 40 + kk] = kp[(size_t)nn * HDm + k0 + kk];
        }
#endif
        __syncthreads();
        const v16bf a = load_a(sA[cur] + wv * 16 * 32, 32, lane);
#pragma unroll
        for (int nt = 0; nt < 4; ++nt) {
            const v16bf bm = load_bt(sBT[cur] + nt * 16 * 40, 40, lane);
            acc[nt] = wmma_bf16(a, bm, acc[nt]);
        }
        __syncthreads();
    }
    const float scale = 1.0f / 16.0f;  // 1/sqrt(HD=256)
    const int n = lane & 15, mb = (lane >> 4) << 3;
    float* op = tatt + ((size_t)b * Lm + i0 + wv * 16) * Lm + j0;
#pragma unroll
    for (int nt = 0; nt < 4; ++nt)
#pragma unroll
        for (int r = 0; r < 8; ++r)
            op[(size_t)(mb + r) * Lm + nt * 16 + n] = acc[nt][r] * scale;
}

// ---------------------------------------------------------------------------
// Kernel 6: temporal softmax row stats (one wave32 per row, shfl reductions).
// ---------------------------------------------------------------------------
__global__ __launch_bounds__(256) void k_tstats(
    const float* __restrict__ tatt, float* __restrict__ tm,
    float* __restrict__ tinv) {
    const int lane = threadIdx.x & 31, wv = threadIdx.x >> 5;
    const int row  = blockIdx.x * 8 + wv;
    const float* p = tatt + (size_t)row * Lm;
    float m = -3.4e38f;
    for (int j = lane; j < Lm; j += 32) {
        __builtin_prefetch(p + j + 256, 0, 1);
        m = fmaxf(m, p[j]);
    }
#pragma unroll
    for (int off = 16; off > 0; off >>= 1) m = fmaxf(m, __shfl_xor(m, off, 32));
    float s = 0.0f;
    for (int j = lane; j < Lm; j += 32) s += __expf(p[j] - m);
#pragma unroll
    for (int off = 16; off > 0; off >>= 1) s += __shfl_xor(s, off, 32);
    if (lane == 0) { tm[row] = m; tinv[row] = 1.0f / s; }
}

// ---------------------------------------------------------------------------
// Kernel 7: normalize t_att in place (NT final store) and t_att @ V via WMMA.
// V B-tiles (256x32): double-buffered TDM, in flight across softmax + WMMA.
// ---------------------------------------------------------------------------
__global__ __launch_bounds__(256) void k_tatt_av(
    float* __restrict__ tatt, const float* __restrict__ tm,
    const float* __restrict__ tinv, const __bf16* __restrict__ VbT,
    float* __restrict__ tout, __bf16* __restrict__ toutb) {
    __shared__ __attribute__((aligned(16))) __bf16 sAtt[128 * 32];
    __shared__ __attribute__((aligned(16))) __bf16 sVT[2][256 * 40];
    __shared__ float s_m[128], s_inv[128];
    const int tid = threadIdx.x, lane = tid & 31, wv = tid >> 5;
    const int b  = blockIdx.x >> 3;
    const int i0 = (blockIdx.x & 7) * 128;

    if (tid < 128) {
        s_m[tid]   = tm[(size_t)b * Lm + i0 + tid];
        s_inv[tid] = tinv[(size_t)b * Lm + i0 + tid];
    }
    v8f acc[16];
#pragma unroll
    for (int i = 0; i < 16; ++i) acc[i] = zero8();

    float* tp         = tatt + ((size_t)b * Lm + i0) * Lm;
    const __bf16* vpT = VbT + (size_t)b * HDm * Lm;  // [256][1024]
    __syncthreads();

#if HAVE_TDM
    if (wv == 0)  // prologue: j0=0 tile into buffer 0
        tdm_load(make_tdm_2d_bf16(vpT, lds_off_of(sVT[0]), 32, 256, Lm,
                                  true, 3, 3));
#endif
    for (int j0 = 0; j0 < Lm; j0 += 32) {
        const int cur = (j0 >> 5) & 1, nxt = cur ^ 1;
#if HAVE_TDM
        if (wv == 0 && j0 + 32 < Lm)
            tdm_load(make_tdm_2d_bf16(vpT + j0 + 32, lds_off_of(sVT[nxt]),
                                      32, 256, Lm, true, 3, 3));
#endif
        for (int idx = tid; idx < 128 * 32; idx += 256) {
            const int i = idx >> 5, j = idx & 31;
            const float raw = tp[(size_t)i * Lm + j0 + j];
            const float pv  = __expf(raw - s_m[i]) * s_inv[i];
            __builtin_nontemporal_store(pv, &tp[(size_t)i * Lm + j0 + j]);
            sAtt[idx] = (__bf16)pv;
        }
#if HAVE_TDM
        if (wv == 0) {
            if (j0 + 32 < Lm)
                __builtin_amdgcn_s_wait_tensorcnt(1);  // cur done, nxt in flight
            else
                __builtin_amdgcn_s_wait_tensorcnt(0);
        }
#else
        for (int idx = tid; idx < 256 * 32; idx += 256) {
            const int d = idx >> 5, jj = idx & 31;
            sVT[cur][d * 40 + jj] = vpT[(size_t)d * Lm + j0 + jj];
        }
#endif
        __syncthreads();
        const v16bf a = load_a(sAtt + wv * 16 * 32, 32, lane);
#pragma unroll
        for (int nt = 0; nt < 16; ++nt) {
            const v16bf bm = load_bt(sVT[cur] + nt * 16 * 40, 40, lane);
            acc[nt] = wmma_bf16(a, bm, acc[nt]);
        }
        __syncthreads();
    }
    const int n = lane & 15, mb = (lane >> 4) << 3;
    const size_t base = ((size_t)b * Lm + i0 + wv * 16) * HDm;
#pragma unroll
    for (int nt = 0; nt < 16; ++nt)
#pragma unroll
        for (int r = 0; r < 8; ++r) {
            const float v  = acc[nt][r];
            const size_t o = base + (size_t)(mb + r) * HDm + nt * 16 + n;
            __builtin_nontemporal_store(v, &tout[o]);  // output #4
            toutb[o] = (__bf16)v;                      // bf16 for final proj
        }
}

// ---------------------------------------------------------------------------
// Kernel 8: out = temporal_out @ Wfc + bfc   ([16384,256] @ [256,64]).
// ---------------------------------------------------------------------------
__global__ __launch_bounds__(256) void k_out(
    const __bf16* __restrict__ toutb, const float* __restrict__ Wfc,
    const float* __restrict__ bfc, float* __restrict__ out) {
    __shared__ __attribute__((aligned(16))) __bf16 sA[128 * 32];
    __shared__ __attribute__((aligned(16))) __bf16 sBT[64 * 40];
    const int tid = threadIdx.x, lane = tid & 31, wv = tid >> 5;
    const int r0 = blockIdx.x * 128;

    v8f acc[4];
#pragma unroll
    for (int i = 0; i < 4; ++i) acc[i] = zero8();

    for (int k0 = 0; k0 < HDm; k0 += 32) {
        for (int idx = tid; idx < 128 * 32; idx += 256) {
            const int i = idx >> 5, kk = idx & 31;
            sA[idx] = toutb[(size_t)(r0 + i) * HDm + k0 + kk];
        }
        for (int idx = tid; idx < 64 * 32; idx += 256) {
            const int nn = idx & 63, kk = idx >> 6;
            sBT[nn * 40 + kk] = (__bf16)Wfc[(size_t)(k0 + kk) * Dm + nn];
        }
        __syncthreads();
        const v16bf a = load_a(sA + wv * 16 * 32, 32, lane);
#pragma unroll
        for (int nt = 0; nt < 4; ++nt) {
            const v16bf bm = load_bt(sBT + nt * 16 * 40, 40, lane);
            acc[nt] = wmma_bf16(a, bm, acc[nt]);
        }
        __syncthreads();
    }
    const int n = lane & 15, mb = (lane >> 4) << 3;
#pragma unroll
    for (int nt = 0; nt < 4; ++nt) {
        const float bn = bfc[nt * 16 + n];
#pragma unroll
        for (int r = 0; r < 8; ++r)
            __builtin_nontemporal_store(
                acc[nt][r] + bn,
                &out[(size_t)(r0 + wv * 16 + mb + r) * Dm + nt * 16 + n]);
    }
}

// ---------------------------------------------------------------------------
extern "C" void kernel_launch(void* const* d_in, const int* in_sizes, int n_in,
                              void* d_out, int out_size, void* d_ws,
                              size_t ws_size, hipStream_t stream) {
    (void)in_sizes; (void)n_in; (void)out_size; (void)ws_size;
    const float* x   = (const float*)d_in[0];
    const float* W   = (const float*)d_in[1];
    const float* a1  = (const float*)d_in[2];
    const float* a2  = (const float*)d_in[3];
    const float* Wq  = (const float*)d_in[4];
    const float* bq  = (const float*)d_in[5];
    const float* Wk  = (const float*)d_in[6];
    const float* bk  = (const float*)d_in[7];
    const float* Wv  = (const float*)d_in[8];
    const float* bv  = (const float*)d_in[9];
    const float* Wfc = (const float*)d_in[10];
    const float* bfc = (const float*)d_in[11];

    // Output tuple, concatenated flat: out, gat_att, t_att, temporal_out.
    float* out  = (float*)d_out;
    float* gat  = out + (size_t)Bm * Lm * Dm;
    float* tat  = gat + (size_t)Hm * Bm * Lm * Lm;
    float* tout = tat + (size_t)Bm * Lm * Lm;

    // Workspace bump allocator (256-B aligned).
    char* ws = (char*)d_ws;
    size_t off = 0;
    auto alloc = [&](size_t bytes) -> void* {
        void* p = ws + off;
        off += (bytes + 255) & ~(size_t)255;
        return p;
    };
    const size_t HBL = (size_t)Hm * Bm * Lm;
    const size_t BL  = (size_t)Bm * Lm;
    __bf16* WhbT = (__bf16*)alloc(HBL * Dm * sizeof(__bf16));
    float*  si   = (float*)alloc(HBL * sizeof(float));
    float*  sj   = (float*)alloc(HBL * sizeof(float));
    float*  gm   = (float*)alloc(HBL * sizeof(float));
    float*  ginv = (float*)alloc(HBL * sizeof(float));
    float*  go   = (float*)alloc(BL * HDm * sizeof(float));
    __bf16* Qb   = (__bf16*)alloc(BL * HDm * sizeof(__bf16));
    __bf16* Kb   = (__bf16*)alloc(BL * HDm * sizeof(__bf16));
    __bf16* VbT  = (__bf16*)alloc(BL * HDm * sizeof(__bf16));
    float*  tmw  = (float*)alloc(BL * sizeof(float));
    float*  tinv = (float*)alloc(BL * sizeof(float));
    __bf16* toutb = (__bf16*)alloc(BL * HDm * sizeof(__bf16));

    k_wh<<<dim3(Hm * Bm * 8), 256, 0, stream>>>(x, W, a1, a2, WhbT, si, sj);
    k_gat_stats<<<dim3(Hm * Bm), 256, 0, stream>>>(si, sj, gm, ginv);
    k_gat_att<<<dim3(Hm * Bm * 8), 256, 0, stream>>>(si, sj, gm, ginv, WhbT,
                                                     gat, go);
    k_qkv<<<dim3(128 * 4, 1, 3), 256, 0, stream>>>(go, Wq, bq, Wk, bk, Wv, bv,
                                                   Qb, Kb, VbT);
    k_scores<<<dim3(Bm * 8 * 16), 256, 0, stream>>>(Qb, Kb, tat);
    k_tstats<<<dim3(Bm * Lm / 8), 256, 0, stream>>>(tat, tmw, tinv);
    k_tatt_av<<<dim3(Bm * 8), 256, 0, stream>>>(tat, tmw, tinv, VbT, tout,
                                                toutb);
    k_out<<<dim3(Bm * Lm / 128), 256, 0, stream>>>(toutb, Wfc, bfc, out);
}
